// BahdanauInterAttention_53188874993873
// MI455X (gfx1250) — matrix operand order repaired and checked by reading, not compiled, not run on hardware
//
#include <hip/hip_runtime.h>
#include <hip/hip_bf16.h>
#include <math.h>

// ---------------------------------------------------------------------------
// Problem constants
// ---------------------------------------------------------------------------
#define B_SZ   64
#define T_ENC  4096
#define H_ENC  512
#define H_DEC  512
#define ATTN_A 256

// WMMA tiling for enc_proj GEMM: M rows of T, N = ATTN (256), K = H_ENC (512)
#define KSTEPS    (H_ENC / 32)   // 16 k-steps of 32
#define NTILES    (ATTN_A / 16)  // 16 n-tiles of 16
#define WG_MROWS  128            // rows of T per workgroup (8 waves x 16 rows)

typedef __attribute__((ext_vector_type(16))) __bf16 v16bf;
typedef __attribute__((ext_vector_type(8)))  float  v8f;

static __device__ __forceinline__ v8f wmma_bf16(v16bf a, v16bf b, v8f c) {
    return __builtin_amdgcn_wmma_f32_16x16x32_bf16(false, a, false, b,
                                                   (short)0, c, false, false);
}

// ---------------------------------------------------------------------------
// Setup kernel: split W_enc (ATTN x H_ENC fp32, row-major) into bf16 hi/lo,
// stored directly in WMMA B-fragment order:
//   idx = ((kstep*NTILES + ntile)*32 + lane)*16 + j
//   where n = ntile*16 + (lane&15) (B col), k = kstep*32 + (j<8 ? 8*(lane>>4)+j : 16+8*(lane>>4)+j-8) (B row)
//   B[k][n] = W_enc[n][k]
// ---------------------------------------------------------------------------
__global__ __launch_bounds__(256) void k_wsplit(const float* __restrict__ We,
                                                __bf16* __restrict__ Whi,
                                                __bf16* __restrict__ Wlo) {
    int idx = blockIdx.x * 256 + threadIdx.x;            // 0 .. 131071
    int j     =  idx        & 15;
    int lane  = (idx >> 4)  & 31;
    int ntile = (idx >> 9)  & 15;
    int ks    =  idx >> 13;
    int k = ks * 32 + (j < 8 ? 8 * (lane >> 4) + j : 16 + 8 * (lane >> 4) + (j - 8));
    int n = ntile * 16 + (lane & 15);
    float w  = We[(size_t)n * H_ENC + k];
    __bf16 h = (__bf16)w;
    Whi[idx] = h;
    Wlo[idx] = (__bf16)(w - (float)h);
}

// ---------------------------------------------------------------------------
// dec_proj[b][a] = sum_d decoder_state[b][d] * W_dec[a][d]   (64 x 256, tiny)
// ---------------------------------------------------------------------------
__global__ __launch_bounds__(256) void k_decproj(const float* __restrict__ ds,
                                                 const float* __restrict__ Wd,
                                                 float* __restrict__ decp) {
    int idx = blockIdx.x * 256 + threadIdx.x;            // 0 .. 16383
    int b = idx >> 8, a = idx & 255;
    const float* x = ds + (size_t)b * H_DEC;
    const float* w = Wd + (size_t)a * H_DEC;
    float s = 0.f;
    for (int i = 0; i < H_DEC; ++i) s = fmaf(x[i], w[i], s);
    decp[idx] = s;
}

__global__ __launch_bounds__(256) void k_zero_ctx(float* __restrict__ ctx) {
    int idx = blockIdx.x * 256 + threadIdx.x;
    if (idx < B_SZ * H_ENC) ctx[idx] = 0.f;
}

// ---------------------------------------------------------------------------
// Main fused kernel: scores[b][t] = v . tanh(enc[b,t,:] @ W_enc^T + dec_proj[b])
// Split-bf16 fp32 emulation: acc += Ahi*Bhi + Ahi*Blo + Alo*Bhi (f32 accum).
// B fragments staged via double-buffered GLOBAL_LOAD_ASYNC_TO_LDS_B128
// (ASYNCcnt pipeline, one barrier per k-step). Fragment halves stored in
// separate 16B/lane arrays -> 2-way (optimal) LDS banking for ds_load_b128.
//
// NOTE: the n-tile loop MUST be fully unrolled. A partially-unrolled loop
// indexes acc[] with a runtime variable and the compiler demotes the
// accumulators to scratch memory (verified: scratch_load/store_b128 spills).
// Full unroll keeps acc[] in VGPRs; >256 live VGPRs with s_set_vgpr_msb
// addressing is far cheaper than per-iteration scratch traffic.
// ---------------------------------------------------------------------------
__global__ __launch_bounds__(256) void k_scores(const float*  __restrict__ enc,
                                                const __bf16* __restrict__ Whi,
                                                const __bf16* __restrict__ Wlo,
                                                const float*  __restrict__ decp,
                                                const float*  __restrict__ vvec,
                                                float* __restrict__ scores) {
    const int b    = blockIdx.y;
    const int t0   = blockIdx.x * WG_MROWS;
    const int tid  = threadIdx.x;
    const int wave = tid >> 5;
    const int lane = tid & 31;
    const int lm   = lane & 15;     // A: row-in-tile / C: col-in-tile
    const int kh   = lane >> 4;     // A: which k-half / C: which m-half

    // [buf][hi=0/lo=1][frag-half][ntile*32 + lane] : 2*2*2*512*16B = 64 KB
    __shared__ uint4 sB[2][2][2][NTILES * 32];

    // This lane's encoder row (A-matrix row m = lane&15 per ISA layout)
    const float* encRow =
        enc + ((size_t)b * T_ENC + (t0 + wave * 16 + lm)) * H_ENC;

    v8f acc[NTILES];
    #pragma unroll
    for (int n = 0; n < NTILES; ++n)
        acc[n] = v8f{0.f,0.f,0.f,0.f,0.f,0.f,0.f,0.f};

    // ---- async staging of one k-step's B fragments (8 asyncs / wave) ----
    auto stage = [&](int ks, int buf) {
        const uint4* gh = (const uint4*)(Whi + (size_t)ks * (NTILES * 32 * 16));
        const uint4* gl = (const uint4*)(Wlo + (size_t)ks * (NTILES * 32 * 16));
        #pragma unroll
        for (int it = 0; it < 4; ++it) {
            int i     = it * 256 + tid;       // uint4 index, 0..1023
            int half  = i & 1;
            int lane_ = (i >> 1) & 31;
            int n_    = i >> 6;
            unsigned dsth = (unsigned)(uintptr_t)&sB[buf][0][half][n_ * 32 + lane_];
            unsigned dstl = (unsigned)(uintptr_t)&sB[buf][1][half][n_ * 32 + lane_];
            unsigned long long srch = (unsigned long long)(uintptr_t)(gh + i);
            unsigned long long srcl = (unsigned long long)(uintptr_t)(gl + i);
            asm volatile("global_load_async_to_lds_b128 %0, %1, off"
                         :: "v"(dsth), "v"(srch) : "memory");
            asm volatile("global_load_async_to_lds_b128 %0, %1, off"
                         :: "v"(dstl), "v"(srcl) : "memory");
        }
    };

    // ---- A-row chunk loader (fp32, ISA 16x32 bf16 A layout) ----
    auto load_a = [&](float* af, int ks) {
        const float* p0 = encRow + ks * 32 + kh * 8;
        const float* p1 = encRow + ks * 32 + 16 + kh * 8;
        #pragma unroll
        for (int j = 0; j < 8; ++j) { af[j] = p0[j]; af[8 + j] = p1[j]; }
    };

    float af[16];
    load_a(af, 0);
    stage(0, 0);

    for (int ks = 0; ks < KSTEPS; ++ks) {
        const int buf = ks & 1;

        // drain our async copies for buf, then sync: data visible everywhere
        // AND every wave is done reading buf^1 from the previous iteration.
        asm volatile("s_wait_asynccnt 0x0" ::: "memory");
        __syncthreads();

        if (ks + 1 < KSTEPS) stage(ks + 1, buf ^ 1);   // overlap with compute

        // split current A chunk to bf16 hi/lo
        v16bf ahi, alo;
        #pragma unroll
        for (int j = 0; j < 16; ++j) {
            __bf16 h = (__bf16)af[j];
            ahi[j] = h;
            alo[j] = (__bf16)(af[j] - (float)h);
        }
        // prefetch next A chunk into registers while WMMAs run
        float afn[16];
        if (ks + 1 < KSTEPS) load_a(afn, ks + 1);

        #pragma unroll
        for (int n = 0; n < NTILES; ++n) {
            union { uint4 q[2]; v16bf v; } bh, bl;
            bh.q[0] = sB[buf][0][0][n * 32 + lane];
            bh.q[1] = sB[buf][0][1][n * 32 + lane];
            bl.q[0] = sB[buf][1][0][n * 32 + lane];
            bl.q[1] = sB[buf][1][1][n * 32 + lane];
            acc[n] = wmma_bf16(ahi, bh.v, acc[n]);
            acc[n] = wmma_bf16(ahi, bl.v, acc[n]);
            acc[n] = wmma_bf16(alo, bh.v, acc[n]);
        }

        if (ks + 1 < KSTEPS) {
            #pragma unroll
            for (int j = 0; j < 16; ++j) af[j] = afn[j];
        }
    }

    // ---- epilogue: score[m] = sum_a tanh(proj[m][a] + dec[a]) * v[a] ----
    // C-tile layout: lane -> N = lane&15 ; VGPR i -> M = i + 8*(lane>>4)
    float part[8];
    #pragma unroll
    for (int i = 0; i < 8; ++i) part[i] = 0.f;
    #pragma unroll
    for (int n = 0; n < NTILES; ++n) {
        int   col = n * 16 + lm;
        float d   = decp[b * ATTN_A + col];
        float vv  = vvec[col];
        #pragma unroll
        for (int i = 0; i < 8; ++i)
            part[i] += tanhf(acc[n][i] + d) * vv;
    }
    // reduce across the 16 lanes of each half-wave (masks 1,2,4,8 stay inside)
    #pragma unroll
    for (int i = 0; i < 8; ++i) {
        float s = part[i];
        s += __shfl_xor(s, 1);
        s += __shfl_xor(s, 2);
        s += __shfl_xor(s, 4);
        s += __shfl_xor(s, 8);
        if (lm == 0)
            scores[(size_t)b * T_ENC + t0 + wave * 16 + kh * 8 + i] = s;
    }
}

// ---------------------------------------------------------------------------
// Softmax over T per batch: one workgroup (256 threads) per b.
// ---------------------------------------------------------------------------
__global__ __launch_bounds__(256) void k_softmax(const float* __restrict__ scores,
                                                 float* __restrict__ attn) {
    const int b = blockIdx.x;
    const float* s = scores + (size_t)b * T_ENC;
    __shared__ float red[8];

    float m = -INFINITY;
    for (int t = threadIdx.x; t < T_ENC; t += 256) m = fmaxf(m, s[t]);
    for (int off = 16; off >= 1; off >>= 1) m = fmaxf(m, __shfl_xor(m, off));
    if ((threadIdx.x & 31) == 0) red[threadIdx.x >> 5] = m;
    __syncthreads();
    float mm = red[0];
    #pragma unroll
    for (int i = 1; i < 8; ++i) mm = fmaxf(mm, red[i]);
    __syncthreads();

    float sum = 0.f;
    for (int t = threadIdx.x; t < T_ENC; t += 256) sum += expf(s[t] - mm);
    for (int off = 16; off >= 1; off >>= 1) sum += __shfl_xor(sum, off);
    if ((threadIdx.x & 31) == 0) red[threadIdx.x >> 5] = sum;
    __syncthreads();
    float tot = 0.f;
    #pragma unroll
    for (int i = 0; i < 8; ++i) tot += red[i];
    float inv = 1.f / tot;

    for (int t = threadIdx.x; t < T_ENC; t += 256)
        attn[(size_t)b * T_ENC + t] = expf(s[t] - mm) * inv;
}

// ---------------------------------------------------------------------------
// context[b][e] = sum_t enc[b][t][e] * attn[b][t]    (memory-bound, 512 MB)
// Grid: (t-chunks=4, e-blocks=2, B). One fp32 atomic per thread at the end.
// ---------------------------------------------------------------------------
__global__ __launch_bounds__(256) void k_context(const float* __restrict__ enc,
                                                 const float* __restrict__ attn,
                                                 float* __restrict__ ctx) {
    const int b  = blockIdx.z;
    const int e  = blockIdx.y * 256 + threadIdx.x;      // 0..511
    const int tc = blockIdx.x;                          // 0..3 (1024 t each)
    const float* E = enc + ((size_t)b * T_ENC + tc * 1024) * H_ENC + e;
    const float* A = attn + (size_t)b * T_ENC + tc * 1024;
    float s = 0.f;
    for (int t = 0; t < 1024; ++t) s = fmaf(E[(size_t)t * H_ENC], A[t], s);
    atomicAdd(&ctx[(size_t)b * H_ENC + e], s);
}

// ---------------------------------------------------------------------------
// Launch
// ---------------------------------------------------------------------------
extern "C" void kernel_launch(void* const* d_in, const int* in_sizes, int n_in,
                              void* d_out, int out_size, void* d_ws, size_t ws_size,
                              hipStream_t stream) {
    const float* decoder_state = (const float*)d_in[0];   // (64, 512)
    const float* encoder_out   = (const float*)d_in[1];   // (64, 4096, 512)
    const float* W_enc         = (const float*)d_in[2];   // (256, 512)
    const float* W_dec         = (const float*)d_in[3];   // (256, 512)
    const float* v             = (const float*)d_in[4];   // (1, 256)

    float* ctx_out  = (float*)d_out;                      // (64, 512)
    float* attn_out = ctx_out + (size_t)B_SZ * H_ENC;     // (64, 4096)

    // Workspace layout
    char*   ws     = (char*)d_ws;
    float*  decp   = (float*)(ws);                                        // 64 KB
    float*  scores = (float*)(ws + 64 * 1024);                            // 1 MB
    __bf16* Whi    = (__bf16*)(ws + 64 * 1024 + 1024 * 1024);             // 256 KB
    __bf16* Wlo    = (__bf16*)(ws + 64 * 1024 + 1024 * 1024 + 256 * 1024);// 256 KB

    // Setup
    k_zero_ctx<<<dim3((B_SZ * H_ENC + 255) / 256), dim3(256), 0, stream>>>(ctx_out);
    k_wsplit  <<<dim3(512), dim3(256), 0, stream>>>(W_enc, Whi, Wlo);
    k_decproj <<<dim3(64),  dim3(256), 0, stream>>>(decoder_state, W_dec, decp);

    // Fused projection + tanh + v-dot (WMMA split-bf16, async LDS pipeline)
    k_scores<<<dim3(T_ENC / WG_MROWS, B_SZ), dim3(256), 0, stream>>>(
        encoder_out, Whi, Wlo, decp, v, scores);

    // Softmax and context
    k_softmax<<<dim3(B_SZ), dim3(256), 0, stream>>>(scores, attn_out);
    k_context<<<dim3(4, H_ENC / 256, B_SZ), dim3(256), 0, stream>>>(
        encoder_out, attn_out, ctx_out);
}
